// Model_31696858644738
// MI455X (gfx1250) — matrix operand order repaired and checked by reading, not compiled, not run on hardware
//
#include <hip/hip_runtime.h>
#include <hip/hip_bf16.h>

typedef __attribute__((ext_vector_type(16))) _Float16 v16h;
typedef __attribute__((ext_vector_type(8)))  float    v8f;
typedef __attribute__((ext_vector_type(4)))  int      v4i;

#define D_MODEL 512
#define HID     2048
#define B_      8
#define N_      207
#define Q_      96
#define S_      72
#define SP_     96          // S padded to multiple of 32 for WMMA K
#define L_      864
#define PATCH_  12
#define ROWS    (B_*N_*Q_)                 // 158976
#define ATTN_ELEMS ((size_t)B_*N_*Q_*S_)   // 11446272

// ---- gfx1250 async global->LDS path (guarded: falls back to uint4 copies) ----
#if defined(__has_builtin)
#  if __has_builtin(__builtin_amdgcn_global_load_async_to_lds_b128)
#    define ASYNC_LDS 1
#  endif
#  if __has_builtin(__builtin_amdgcn_s_wait_asynccnt)
#    define HAVE_WAIT_ASYNC 1
#  endif
#endif

#ifdef ASYNC_LDS
typedef __attribute__((address_space(1))) v4i GV4;   // global (prints as __device__)
typedef __attribute__((address_space(3))) v4i LV4;   // LDS    (prints as __shared__)
__device__ __forceinline__ void async_cp16(const void* g, void* l) {
    __builtin_amdgcn_global_load_async_to_lds_b128((GV4*)g, (LV4*)l, 0, 0);
}
__device__ __forceinline__ void wait_async0() {
#  ifdef HAVE_WAIT_ASYNC
    __builtin_amdgcn_s_wait_asynccnt(0);
#  else
    asm volatile("s_wait_asynccnt 0" ::: "memory");
#  endif
}
#else
__device__ __forceinline__ void async_cp16(const void* g, void* l) {
    *(uint4*)l = *(const uint4*)g;
}
__device__ __forceinline__ void wait_async0() {}
#endif

// ---------------- WMMA fragment loaders ----------------
// A operand (16x32 f16): lane m%16 = row, lane/16 selects K-half.
// elems 0..15 = K(base..base+15) (K-contiguous, base includes half*16).
__device__ __forceinline__ v16h load_a_frag(const _Float16* rowbase) {
    v16h a;
    const uint4* p = (const uint4*)rowbase;
    ((uint4*)&a)[0] = p[0];
    ((uint4*)&a)[1] = p[1];
    return a;
}
// B operand (32x16 f16): per-lane 16 contiguous halves (K-contiguous layout).
__device__ __forceinline__ v16h load_b_frag(const _Float16* base) {
    v16h b;
    const uint4* p = (const uint4*)base;
    ((uint4*)&b)[0] = p[0];
    ((uint4*)&b)[1] = p[1];
    return b;
}
#define WMMA(a, b, c) __builtin_amdgcn_wmma_f32_16x16x32_f16(false, (a), false, (b), (short)0, (c), false, false)

// ---------------- Kernel 0: swizzle weights into WMMA B-fragment order ----
// Ws flat = ((ntile*nKsteps + ks)*32 + lane)*16 + e
//   value W[k*Nfull + n], k = ks*32 + e + 16*(lane>=16), n = ntile*16 + (lane&15)
__global__ __launch_bounds__(256) void prep_w(const float* __restrict__ W,
                                              _Float16* __restrict__ Ws,
                                              int Nfull, int nKsteps) {
    int idx = blockIdx.x * 256 + threadIdx.x;
    int e    = idx & 15;
    int lane = (idx >> 4) & 31;
    int rest = idx >> 9;
    int ks   = rest % nKsteps;
    int nt   = rest / nKsteps;
    int k = ks * 32 + e + ((lane >> 4) << 4);
    int n = nt * 16 + (lane & 15);
    Ws[idx] = (_Float16)W[k * Nfull + n];
}

// ---------------- Kernel 1: patch embedding -> encT f16 [B*N][512][72] ----
// one workgroup per (b,n): d-major output so attention's B operand is a bulk copy
__global__ __launch_bounds__(256) void patch_embed(const float* __restrict__ tgt,
                                                   const float* __restrict__ cw,
                                                   const float* __restrict__ cb,
                                                   _Float16* __restrict__ encT) {
    __shared__ float xp[L_];                // 864 floats
    __shared__ float scw[D_MODEL * PATCH_]; // 6144 floats (24 KB)
    int tid = threadIdx.x;
    int bn = blockIdx.x;
    int n  = bn % N_;
    int b  = bn / N_;
    for (int i = tid; i < L_; i += 256)
        xp[i] = tgt[((size_t)b * L_ + i) * N_ + n];
    for (int i = tid; i < D_MODEL * PATCH_; i += 256)
        scw[i] = cw[i];
    __syncthreads();
    _Float16* dst = encT + (size_t)bn * D_MODEL * S_;
    for (int idx = tid; idx < D_MODEL * S_; idx += 256) {
        int d = idx / S_, s = idx % S_;
        float acc = cb[d];
        #pragma unroll
        for (int k = 0; k < PATCH_; ++k) acc += xp[s * PATCH_ + k] * scw[d * PATCH_ + k];
        dst[idx] = (_Float16)acc;
    }
}

// ---------------- Kernel 2: cosine attention + ct = attn @ enc (WMMA) -----
// one workgroup per (b,n); 8 waves; 4 N-tiles register-blocked per A fragment
__global__ __launch_bounds__(256) void attention(const float* __restrict__ enc_time,
                                                 const float* __restrict__ dec_time,
                                                 const _Float16* __restrict__ encT,
                                                 float* __restrict__ attn_out,
                                                 _Float16* __restrict__ ctH) {
    extern __shared__ char smem[];
    float*    sEt   = (float*)smem;                       // 72*4
    float*    sDt   = sEt + S_ * 4;                       // 96*4
    float*    sNe   = sDt + Q_ * 4;                       // 72
    float*    sNd   = sNe + S_;                           // 96
    _Float16* sAttn = (_Float16*)(sNd + Q_);              // 96*96 halves
    _Float16* sEncT = (_Float16*)((char*)sAttn + Q_ * SP_ * 2); // 512*96 halves
    float*    sSim  = (float*)sEncT;                      // alias (96*72 f32 fits)

    int tid  = threadIdx.x;
    int bn   = blockIdx.x;
    int n    = bn % N_;
    int b    = bn / N_;

    // time features
    for (int i = tid; i < S_ * 4; i += 256) {
        int s = i >> 2, c = i & 3;
        sEt[i] = enc_time[(((size_t)b * S_ + s) * N_ + n) * 4 + c];
    }
    for (int i = tid; i < Q_ * 4; i += 256) {
        int q = i >> 2, c = i & 3;
        sDt[i] = dec_time[(((size_t)b * Q_ + q) * N_ + n) * 4 + c];
    }
    __syncthreads();
    if (tid < S_) {
        float x0 = sEt[tid*4], x1 = sEt[tid*4+1], x2 = sEt[tid*4+2], x3 = sEt[tid*4+3];
        sNe[tid] = sqrtf(x0*x0 + x1*x1 + x2*x2 + x3*x3);
    }
    if (tid < Q_) {
        float x0 = sDt[tid*4], x1 = sDt[tid*4+1], x2 = sDt[tid*4+2], x3 = sDt[tid*4+3];
        sNd[tid] = sqrtf(x0*x0 + x1*x1 + x2*x2 + x3*x3);
    }
    __syncthreads();

    // cosine similarity
    for (int i = tid; i < Q_ * S_; i += 256) {
        int q = i / S_, s = i % S_;
        float d0 = sDt[q*4]*sEt[s*4] + sDt[q*4+1]*sEt[s*4+1]
                 + sDt[q*4+2]*sEt[s*4+2] + sDt[q*4+3]*sEt[s*4+3];
        sSim[i] = d0 / (sNd[q] * sNe[s]);
    }
    __syncthreads();

    // softmax rows; attn f32 -> d_out, f16 (K-padded) -> LDS
    if (tid < Q_) {
        int q = tid;
        float mx = -1e30f;
        for (int s = 0; s < S_; ++s) mx = fmaxf(mx, sSim[q*S_ + s]);
        float sum = 0.f;
        for (int s = 0; s < S_; ++s) { float e = __expf(sSim[q*S_+s] - mx); sSim[q*S_+s] = e; sum += e; }
        float inv = 1.f / sum;
        float* arow = attn_out + ((size_t)bn * Q_ + q) * S_;
        for (int s = 0; s < S_; ++s) {
            float a = sSim[q*S_+s] * inv;
            arow[s] = a;
            sAttn[q*SP_ + s] = (_Float16)a;
        }
        for (int s = S_; s < SP_; ++s) sAttn[q*SP_ + s] = (_Float16)0.f;
    }
    __syncthreads();

    // stage enc (already d-major): 512 rows x 144B bulk copies + zero pad cols 72..95
    {
        const _Float16* src = encT + (size_t)bn * D_MODEL * S_;
        for (int i = tid; i < D_MODEL * 9; i += 256) {
            int d = i / 9, j = i - d * 9;
            async_cp16(src + d * S_ + j * 8, sEncT + d * SP_ + j * 8);
        }
        uint4 z = {0u, 0u, 0u, 0u};
        for (int i = tid; i < D_MODEL * 3; i += 256) {
            int d = i / 3, j = i - d * 3;
            *(uint4*)(sEncT + d * SP_ + S_ + j * 8) = z;
        }
        wait_async0();
    }
    __syncthreads();

    // ct = attn(96x96) @ enc(96x512); 48 groups of 4 N-tiles
    int wave = tid >> 5, lane = tid & 31;
    int half = lane >> 4;
    int mrow = lane & 15;
    for (int g = wave; g < 48; g += 8) {
        int mi  = g >> 3;          // 0..5
        int nt0 = (g & 7) * 4;     // 0..28
        v8f a0 = {}, a1 = {}, a2 = {}, a3 = {};
        #pragma unroll
        for (int ks = 0; ks < 3; ++ks) {
            v16h a = load_a_frag(&sAttn[(mi*16 + mrow) * SP_ + ks*32 + half*16]);
            v16h b0 = load_b_frag(&sEncT[((nt0+0)*16 + mrow) * SP_ + ks*32 + half*16]);
            v16h b1 = load_b_frag(&sEncT[((nt0+1)*16 + mrow) * SP_ + ks*32 + half*16]);
            v16h b2 = load_b_frag(&sEncT[((nt0+2)*16 + mrow) * SP_ + ks*32 + half*16]);
            v16h b3 = load_b_frag(&sEncT[((nt0+3)*16 + mrow) * SP_ + ks*32 + half*16]);
            a0 = WMMA(a, b0, a0); a1 = WMMA(a, b1, a1);
            a2 = WMMA(a, b2, a2); a3 = WMMA(a, b3, a3);
        }
        v8f accs[4] = {a0, a1, a2, a3};
        #pragma unroll
        for (int u = 0; u < 4; ++u) {
            int d = (nt0+u)*16 + mrow;
            #pragma unroll
            for (int v = 0; v < 8; ++v) {
                int q = mi*16 + v + 8*half;
                ctH[((size_t)bn * Q_ + q) * D_MODEL + d] = (_Float16)accs[u][v];
            }
        }
    }
}

// ---------------- Kernel 3: fused FFN + residual + LayerNorm + proj -------
// 32-row tile; whole 32x2048 hidden in LDS (CDNA5 320KB/WGP); 4-way N blocking
#define MT 32
#define YP 516   // padded f32 pitch to break LDS bank alignment
__global__ __launch_bounds__(256) void ffn_fused(const _Float16* __restrict__ ctH,
                                                 const _Float16* __restrict__ W1s,
                                                 const float* __restrict__ b1,
                                                 const _Float16* __restrict__ W2s,
                                                 const float* __restrict__ b2,
                                                 const float* __restrict__ ln_g,
                                                 const float* __restrict__ ln_b,
                                                 const float* __restrict__ Wo,
                                                 const float* __restrict__ bo,
                                                 float* __restrict__ outp) {
    extern __shared__ char smem[];
    _Float16* sCt = (_Float16*)smem;                 // 32*512  (32 KB)
    _Float16* sH  = sCt + MT * D_MODEL;              // 32*2048 (128 KB)
    float*    sY  = (float*)(sH + MT * HID);         // 32*516  (66 KB)

    int tid  = threadIdx.x;
    int wave = tid >> 5, lane = tid & 31;
    int half = lane >> 4;
    int mrow = lane & 15;
    size_t R0 = (size_t)blockIdx.x * MT;

    // stage ct tile (async where available)
    {
        const _Float16* src = ctH + R0 * D_MODEL;
        for (int i = tid; i < MT * D_MODEL / 8; i += 256)
            async_cp16(src + i * 8, sCt + i * 8);
        wait_async0();
    }
    __syncthreads();

    // GEMM1: H = relu(ct @ W1 + b1)   [2 mi x 32 groups of 4 N-tiles, K=512]
    for (int g = wave; g < 64; g += 8) {
        int mi  = g >> 5;
        int nt0 = (g & 31) * 4;
        v8f a0 = {}, a1 = {}, a2 = {}, a3 = {};
        for (int ks = 0; ks < D_MODEL/32; ++ks) {
            v16h a = load_a_frag(&sCt[(mi*16 + mrow) * D_MODEL + ks*32 + half*16]);
            v16h b0f = load_b_frag(W1s + ((size_t)((nt0+0)*(D_MODEL/32) + ks)*32 + lane)*16);
            v16h b1f = load_b_frag(W1s + ((size_t)((nt0+1)*(D_MODEL/32) + ks)*32 + lane)*16);
            v16h b2f = load_b_frag(W1s + ((size_t)((nt0+2)*(D_MODEL/32) + ks)*32 + lane)*16);
            v16h b3f = load_b_frag(W1s + ((size_t)((nt0+3)*(D_MODEL/32) + ks)*32 + lane)*16);
            a0 = WMMA(a, b0f, a0); a1 = WMMA(a, b1f, a1);
            a2 = WMMA(a, b2f, a2); a3 = WMMA(a, b3f, a3);
        }
        v8f accs[4] = {a0, a1, a2, a3};
        #pragma unroll
        for (int u = 0; u < 4; ++u) {
            int nn = (nt0+u)*16 + mrow;
            float bias = b1[nn];
            #pragma unroll
            for (int v = 0; v < 8; ++v) {
                int m = mi*16 + v + 8*half;
                sH[m * HID + nn] = (_Float16)fmaxf(accs[u][v] + bias, 0.f);
            }
        }
    }
    __syncthreads();

    // GEMM2: Y = H @ W2 + b2 + ct    [2 mi x 8 groups of 4 N-tiles, K=2048]
    for (int g = wave; g < 16; g += 8) {
        int mi  = g >> 3;
        int nt0 = (g & 7) * 4;
        v8f a0 = {}, a1 = {}, a2 = {}, a3 = {};
        for (int ks = 0; ks < HID/32; ++ks) {
            v16h a = load_a_frag(&sH[(mi*16 + mrow) * HID + ks*32 + half*16]);
            v16h b0f = load_b_frag(W2s + ((size_t)((nt0+0)*(HID/32) + ks)*32 + lane)*16);
            v16h b1f = load_b_frag(W2s + ((size_t)((nt0+1)*(HID/32) + ks)*32 + lane)*16);
            v16h b2f = load_b_frag(W2s + ((size_t)((nt0+2)*(HID/32) + ks)*32 + lane)*16);
            v16h b3f = load_b_frag(W2s + ((size_t)((nt0+3)*(HID/32) + ks)*32 + lane)*16);
            a0 = WMMA(a, b0f, a0); a1 = WMMA(a, b1f, a1);
            a2 = WMMA(a, b2f, a2); a3 = WMMA(a, b3f, a3);
        }
        v8f accs[4] = {a0, a1, a2, a3};
        #pragma unroll
        for (int u = 0; u < 4; ++u) {
            int nn = (nt0+u)*16 + mrow;
            float bias = b2[nn];
            #pragma unroll
            for (int v = 0; v < 8; ++v) {
                int m = mi*16 + v + 8*half;
                sY[m * YP + nn] = accs[u][v] + bias + (float)sCt[m * D_MODEL + nn];
            }
        }
    }
    __syncthreads();

    // LayerNorm + projection: 8 lanes/row, interleaved columns, wave32 shuffles
    {
        int r   = wave * 4 + (lane >> 3);
        int sub = lane & 7;
        float sum = 0.f;
        for (int i = 0; i < 64; ++i) sum += sY[r * YP + sub + 8*i];
        sum += __shfl_xor(sum, 1); sum += __shfl_xor(sum, 2); sum += __shfl_xor(sum, 4);
        float mean = sum * (1.f / D_MODEL);
        float vs = 0.f;
        for (int i = 0; i < 64; ++i) { float d = sY[r * YP + sub + 8*i] - mean; vs += d*d; }
        vs += __shfl_xor(vs, 1); vs += __shfl_xor(vs, 2); vs += __shfl_xor(vs, 4);
        float rstd = rsqrtf(vs * (1.f / D_MODEL) + 1e-5f);
        float dot = 0.f;
        for (int i = 0; i < 64; ++i) {
            int c = sub + 8*i;
            float yn = (sY[r * YP + c] - mean) * rstd * ln_g[c] + ln_b[c];
            dot += yn * Wo[c];
        }
        dot += __shfl_xor(dot, 1); dot += __shfl_xor(dot, 2); dot += __shfl_xor(dot, 4);
        if (sub == 0) {
            size_t R = R0 + r;                 // R = (b*N + n)*Q + q
            int q = (int)(R % Q_);
            int nidx = (int)((R / Q_) % N_);
            int b = (int)(R / ((size_t)Q_ * N_));
            outp[((size_t)b * Q_ + q) * N_ + nidx] = dot + bo[0];
        }
    }
}

extern "C" void kernel_launch(void* const* d_in, const int* in_sizes, int n_in,
                              void* d_out, int out_size, void* d_ws, size_t ws_size,
                              hipStream_t stream) {
    const float* enc_time = (const float*)d_in[0];
    const float* enc_tgt  = (const float*)d_in[1];
    const float* dec_time = (const float*)d_in[2];
    const float* conv_w   = (const float*)d_in[3];
    const float* conv_b   = (const float*)d_in[4];
    const float* W1       = (const float*)d_in[5];
    const float* b1       = (const float*)d_in[6];
    const float* W2       = (const float*)d_in[7];
    const float* b2       = (const float*)d_in[8];
    const float* ln_g     = (const float*)d_in[9];
    const float* ln_b     = (const float*)d_in[10];
    const float* Wo       = (const float*)d_in[11];
    const float* bo       = (const float*)d_in[12];
    float* out = (float*)d_out;

    char* ws = (char*)d_ws;
    _Float16* encT = (_Float16*)ws;                               // 122,093,568 B
    _Float16* ctH  = (_Float16*)(ws + 122093568);                 // 162,791,424 B
    _Float16* W1s  = (_Float16*)(ws + 122093568 + 162791424);     //   2,097,152 B
    _Float16* W2s  = (_Float16*)(ws + 122093568 + 162791424 + 2097152);

    prep_w<<<4096, 256, 0, stream>>>(W1, W1s, HID,     D_MODEL/32);
    prep_w<<<4096, 256, 0, stream>>>(W2, W2s, D_MODEL, HID/32);

    patch_embed<<<B_*N_, 256, 0, stream>>>(enc_tgt, conv_w, conv_b, encT);

    size_t attn_smem = 3360 + (size_t)Q_*SP_*2 + (size_t)D_MODEL*SP_*2;   // ~120 KB
    attention<<<B_*N_, 256, attn_smem, stream>>>(enc_time, dec_time, encT, out, ctH);

    size_t ffn_smem = (size_t)MT*D_MODEL*2 + (size_t)MT*HID*2 + (size_t)MT*YP*4; // ~226 KB
    ffn_fused<<<ROWS/MT, 256, ffn_smem, stream>>>(ctH, W1s, b1, W2s, b2,
                                                  ln_g, ln_b, Wo, bo,
                                                  out + ATTN_ELEMS);
}
